// S4Net_90941637525628
// MI455X (gfx1250) — compile-verified, matrix operand order checked by reading, not run on hardware
//
#include <hip/hip_runtime.h>
#include <hip/hip_bf16.h>
#include <cstdint>

// ---------------- types ----------------
typedef _Float16 h16;
typedef __attribute__((ext_vector_type(8)))  _Float16 v8h;
typedef __attribute__((ext_vector_type(16))) _Float16 v16h;
typedef __attribute__((ext_vector_type(8)))  float    v8f;

#define LYR 2
#define HD  512
#define NHM 32
#define BB  8
#define LL  4096
#define MROWS (BB * LL)   // 32768
#define ATP  520          // padded LDS row stride in halves (1040B = 4 dwords mod 64 banks)

#if __has_builtin(__builtin_amdgcn_global_load_async_to_lds_b128)
#define HAVE_ASYNC_LDS 1
typedef int v4i __attribute__((vector_size(16)));        // GCC-style int32x4 (matches builtin proto)
typedef __attribute__((address_space(1))) v4i* as1_v4i;  // global
typedef __attribute__((address_space(3))) v4i* as3_v4i;  // LDS
#else
#define HAVE_ASYNC_LDS 0
#endif

__device__ __forceinline__ float gelu_exact(float x) {
    return 0.5f * x * (1.0f + erff(x * 0.70710678118654752f));
}

// ---------------- SSM coefficient generation ----------------
__global__ void coef_gen(const float* __restrict__ log_dt,
                         const float* __restrict__ A_re, const float* __restrict__ A_im,
                         const float* __restrict__ B_re, const float* __restrict__ B_im,
                         const float* __restrict__ C_re, const float* __restrict__ C_im,
                         float* __restrict__ dAr_o, float* __restrict__ dAi_o,
                         float* __restrict__ cr_o,  float* __restrict__ ci_o) {
    int i = blockIdx.x * 256 + threadIdx.x;          // over LYR*HD*NHM = 32768
    int lh = i / NHM;                                 // l*HD + h
    float dt = __expf(log_dt[lh]);
    float ar = A_re[i], ai = A_im[i];
    float dr = dt * ar, di = dt * ai;
    float e  = __expf(dr);
    float dAr = e * __cosf(di);
    float dAi = e * __sinf(di);
    float nr = dAr - 1.0f, ni = dAi;
    float den = 1.0f / (ar * ar + ai * ai);
    float qr = (nr * ar + ni * ai) * den;
    float qi = (ni * ar - nr * ai) * den;
    float br = B_re[i], bi = B_im[i];
    float dBr = br * qr - bi * qi;
    float dBi = br * qi + bi * qr;
    float crr = C_re[i], cii = C_im[i];
    dAr_o[i] = dAr;  dAi_o[i] = dAi;
    cr_o[i] = crr * dBr - cii * dBi;
    ci_o[i] = crr * dBi + cii * dBr;
}

// ---------------- weight packing: W[l][k][n] f32 -> Wt[l][n][k] f16 ----------------
__global__ void pack_w(const float* __restrict__ W, h16* __restrict__ Wt) {
    int i = blockIdx.x * 256 + threadIdx.x;          // over LYR*HD*HD
    int l = i >> 18;
    int r = i & (HD * HD - 1);
    int k = r >> 9;
    int n = r & (HD - 1);
    Wt[((size_t)l * HD + n) * HD + k] = (h16)W[i];
}

// ---------------- LayerNorm over H=512 ----------------
__global__ __launch_bounds__(256)
void ln512(const float* __restrict__ in, const float* __restrict__ g, const float* __restrict__ b,
           float* __restrict__ of, h16* __restrict__ oh, float* __restrict__ oT) {
    __shared__ float sm[8];
    __shared__ float s_mean, s_rstd;
    const int row = blockIdx.x;                      // b*LL + t
    const int c0 = threadIdx.x, c1 = threadIdx.x + 256;
    const float* p = in + (size_t)row * HD;
    float x0 = p[c0], x1 = p[c1];

    float s = x0 + x1;
    #pragma unroll
    for (int o = 16; o; o >>= 1) s += __shfl_xor(s, o, 32);
    if ((threadIdx.x & 31) == 0) sm[threadIdx.x >> 5] = s;
    __syncthreads();
    if (threadIdx.x == 0) {
        float t = 0.f;
        #pragma unroll
        for (int i = 0; i < 8; i++) t += sm[i];
        s_mean = t * (1.0f / HD);
    }
    __syncthreads();
    float mean = s_mean;
    float d0 = x0 - mean, d1 = x1 - mean;
    float q = d0 * d0 + d1 * d1;
    #pragma unroll
    for (int o = 16; o; o >>= 1) q += __shfl_xor(q, o, 32);
    __syncthreads();
    if ((threadIdx.x & 31) == 0) sm[threadIdx.x >> 5] = q;
    __syncthreads();
    if (threadIdx.x == 0) {
        float t = 0.f;
        #pragma unroll
        for (int i = 0; i < 8; i++) t += sm[i];
        s_rstd = rsqrtf(t * (1.0f / HD) + 1e-5f);
    }
    __syncthreads();
    float rst = s_rstd;

    float y0 = d0 * rst * g[c0] + b[c0];
    float y1 = d1 * rst * g[c1] + b[c1];
    if (of) { of[(size_t)row * HD + c0] = y0; of[(size_t)row * HD + c1] = y1; }
    if (oh) { oh[(size_t)row * HD + c0] = (h16)y0; oh[(size_t)row * HD + c1] = (h16)y1; }
    if (oT) {
        int bb = row >> 12;
        int t  = row & (LL - 1);
        oT[((size_t)(bb * HD + c0)) * LL + t] = y0;
        oT[((size_t)(bb * HD + c1)) * LL + t] = y1;
    }
}

// ---------------- diagonal SSM scan, wave per (b,h), lane per mode, in-place ----------------
__global__ __launch_bounds__(256)
void ssm_scan(float* __restrict__ buf,
              const float* __restrict__ dAr_, const float* __restrict__ dAi_,
              const float* __restrict__ cr_,  const float* __restrict__ ci_,
              int layerOff) {
    const int lane = threadIdx.x & 31;
    const int w = blockIdx.x * 8 + (threadIdx.x >> 5);   // 0 .. BB*HD-1
    const int h = w & (HD - 1);
    float* p = buf + (size_t)w * LL;
    const int ci = layerOff + h * NHM + lane;
    const float dAr = dAr_[ci], dAi = dAi_[ci];
    const float cr = cr_[ci], cim = ci_[ci];
    float sr = 0.f, si = 0.f;
    for (int t0 = 0; t0 < LL; t0 += 32) {
        float u = p[t0 + lane];
        float yout = 0.f;
        #pragma unroll
        for (int j = 0; j < 32; j++) {
            float uj = __shfl(u, j, 32);
            float nsr = fmaf(dAr, sr, fmaf(-dAi, si, uj));
            float nsi = fmaf(dAr, si, dAi * sr);
            sr = nsr; si = nsi;
            float c = fmaf(cr, sr, -cim * si);
            #pragma unroll
            for (int o = 16; o; o >>= 1) c += __shfl_xor(c, o, 32);
            if (lane == j) yout = 2.0f * c;
        }
        p[t0 + lane] = yout;
    }
}

// ---------------- transpose back + D-skip + GeLU -> f16 GEMM input ----------------
__global__ void post_ssm(const float* __restrict__ yT, const float* __restrict__ z,
                         const float* __restrict__ D, h16* __restrict__ yg, int layer) {
    size_t i = (size_t)blockIdx.x * 256 + threadIdx.x;   // over BB*HD*LL
    int t = (int)(i & (LL - 1));
    int w = (int)(i >> 12);
    int h = w & (HD - 1);
    int bb = w >> 9;
    float v = yT[i] + z[((size_t)(bb * LL + t)) * HD + h] * D[layer * HD + h];
    yg[((size_t)(bb * LL + t)) * HD + h] = (h16)gelu_exact(v);
}

// ---------------- WMMA GEMM with async-to-LDS A staging ----------------
// Block: 8 waves share one 32-row A tile staged into LDS (async path on gfx1250);
// each wave owns a distinct 64-wide N slice -> 2x4 v_wmma_f32_16x16x32_f16 tiles.
__global__ __launch_bounds__(256)
void gemm512_wmma(const h16* __restrict__ A, const h16* __restrict__ Bt,
                  const float* __restrict__ bias, const float* __restrict__ resid,
                  float* __restrict__ Cf, h16* __restrict__ Ch, int applyGelu) {
    __shared__ __align__(16) h16 Atile[32 * ATP];        // 32 rows, padded stride
    const int tid  = threadIdx.x;
    const int lane = tid & 31;
    const int mBase = blockIdx.x * 32;
    const int nBase = (tid >> 5) * 64;
    const int l15 = lane & 15;
    const int kc  = (lane >> 4) * 8;

    // ---- stage A tile: 32 rows x 512 halves, 2048 16B-chunks, 256 threads x 8 rounds
    {
        const h16* gA = A + (size_t)mBase * HD;
        #pragma unroll
        for (int r = 0; r < 8; ++r) {
            int c = r * 256 + tid;
            int row = c >> 6;
            int col8 = (c & 63) * 8;
            const h16* gsrc = gA + row * HD + col8;
            h16* ldst = &Atile[row * ATP + col8];
#if HAVE_ASYNC_LDS
            __builtin_amdgcn_global_load_async_to_lds_b128(
                (as1_v4i)(uintptr_t)gsrc, (as3_v4i)(uintptr_t)ldst, 0, 0);
#else
            *(v8h*)ldst = *(const v8h*)gsrc;
#endif
        }
#if HAVE_ASYNC_LDS
#if __has_builtin(__builtin_amdgcn_s_wait_asynccnt)
        __builtin_amdgcn_s_wait_asynccnt(0);
#else
        asm volatile("s_wait_asynccnt 0" ::: "memory");
#endif
#endif
        __syncthreads();
    }

    v8f acc[2][4];
    #pragma unroll
    for (int i = 0; i < 2; i++)
        #pragma unroll
        for (int j = 0; j < 4; j++)
            acc[i][j] = (v8f){0.f,0.f,0.f,0.f,0.f,0.f,0.f,0.f};

    const h16* aRow0 = &Atile[(l15)      * ATP + kc];    // LDS fragment rows
    const h16* aRow1 = &Atile[(16 + l15) * ATP + kc];
    const h16* bRow[4];
    #pragma unroll
    for (int j = 0; j < 4; j++) bRow[j] = Bt + (size_t)(nBase + j * 16 + l15) * HD + kc;

    for (int kt = 0; kt < HD; kt += 32) {
        if (kt + 32 < HD) {
            __builtin_prefetch((const void*)(bRow[0] + kt + 32), 0, 3);
            __builtin_prefetch((const void*)(bRow[2] + kt + 32), 0, 3);
        }
        v16h af[2], bf[4];
        {
            v8h lo = *(const v8h*)(aRow0 + kt);
            v8h hi = *(const v8h*)(aRow0 + kt + 16);
            af[0] = __builtin_shufflevector(lo, hi, 0,1,2,3,4,5,6,7,8,9,10,11,12,13,14,15);
            lo = *(const v8h*)(aRow1 + kt);
            hi = *(const v8h*)(aRow1 + kt + 16);
            af[1] = __builtin_shufflevector(lo, hi, 0,1,2,3,4,5,6,7,8,9,10,11,12,13,14,15);
        }
        #pragma unroll
        for (int j = 0; j < 4; j++) {
            v8h lo = *(const v8h*)(bRow[j] + kt);
            v8h hi = *(const v8h*)(bRow[j] + kt + 16);
            bf[j] = __builtin_shufflevector(lo, hi, 0,1,2,3,4,5,6,7,8,9,10,11,12,13,14,15);
        }
        #pragma unroll
        for (int i = 0; i < 2; i++)
            #pragma unroll
            for (int j = 0; j < 4; j++)
                acc[i][j] = __builtin_amdgcn_wmma_f32_16x16x32_f16(
                    false, af[i], false, bf[j], (short)0, acc[i][j], false, false);
    }

    // epilogue
    #pragma unroll
    for (int i = 0; i < 2; i++) {
        const int m0 = mBase + i * 16 + (lane >> 4) * 8;
        #pragma unroll
        for (int j = 0; j < 4; j++) {
            const int n = nBase + j * 16 + l15;
            const float bv = bias[n];
            #pragma unroll
            for (int r = 0; r < 8; r++) {
                const size_t idx = (size_t)(m0 + r) * HD + n;
                float v = acc[i][j][r] + bv;
                if (resid) v += resid[idx];
                if (applyGelu) v = gelu_exact(v);
                if (Cf) Cf[idx] = v;
                if (Ch) Ch[idx] = (h16)v;
            }
        }
    }
}

// ---------------- host launcher ----------------
extern "C" void kernel_launch(void* const* d_in, const int* in_sizes, int n_in,
                              void* d_out, int out_size, void* d_ws, size_t ws_size,
                              hipStream_t stream) {
    const float* x     = (const float*)d_in[0];
    const float* ln1w  = (const float*)d_in[1];
    const float* ln1b  = (const float*)d_in[2];
    const float* ln2w  = (const float*)d_in[3];
    const float* ln2b  = (const float*)d_in[4];
    const float* logdt = (const float*)d_in[5];
    const float* A_re  = (const float*)d_in[6];
    const float* A_im  = (const float*)d_in[7];
    const float* B_re  = (const float*)d_in[8];
    const float* B_im  = (const float*)d_in[9];
    const float* C_re  = (const float*)d_in[10];
    const float* C_im  = (const float*)d_in[11];
    const float* Dp    = (const float*)d_in[12];
    const float* Wk    = (const float*)d_in[13];
    const float* bk    = (const float*)d_in[14];
    const float* W1    = (const float*)d_in[15];
    const float* b1    = (const float*)d_in[16];
    const float* W2    = (const float*)d_in[17];
    const float* b2    = (const float*)d_in[18];
    const float* Wout  = (const float*)d_in[19];
    const float* bout  = (const float*)d_in[20];
    const float* Wproj = (const float*)d_in[21];
    const float* bproj = (const float*)d_in[22];
    float* out = (float*)d_out;

    const size_t MH = (size_t)MROWS * HD;
    const size_t WE = (size_t)LYR * HD * HD;
    const size_t CE = (size_t)LYR * HD * NHM;

    char* ws = (char*)d_ws;
    float* P1  = (float*)ws;              ws += MH * 4;      // z / o1 (f32)
    float* P2  = (float*)ws;              ws += MH * 4;      // zT -> yT / proj tmp (f32)
    h16*  Q0   = (h16*)ws;                ws += MH * 2;      // yg / o1 f16 / h
    h16*  Q1   = (h16*)ws;                ws += MH * 2;      // y1 / z2
    h16*  WkT  = (h16*)ws;                ws += WE * 2;
    h16*  W1T  = (h16*)ws;                ws += WE * 2;
    h16*  W2T  = (h16*)ws;                ws += WE * 2;
    h16*  WoT  = (h16*)ws;                ws += WE * 2;
    h16*  WpT  = (h16*)ws;                ws += WE * 2;
    float* dAr = (float*)ws;              ws += CE * 4;
    float* dAi = (float*)ws;              ws += CE * 4;
    float* cfr = (float*)ws;              ws += CE * 4;
    float* cfi = (float*)ws;              ws += CE * 4;

    coef_gen<<<CE / 256, 256, 0, stream>>>(logdt, A_re, A_im, B_re, B_im, C_re, C_im,
                                           dAr, dAi, cfr, cfi);
    pack_w<<<WE / 256, 256, 0, stream>>>(Wk,    WkT);
    pack_w<<<WE / 256, 256, 0, stream>>>(W1,    W1T);
    pack_w<<<WE / 256, 256, 0, stream>>>(W2,    W2T);
    pack_w<<<WE / 256, 256, 0, stream>>>(Wout,  WoT);
    pack_w<<<WE / 256, 256, 0, stream>>>(Wproj, WpT);

    const int gemmBlocks = MROWS / 32;                     // 1024 (one A tile per block)
    const int scanBlocks = (BB * HD) / 8;                  // 512
    const int postBlocks = (int)((size_t)BB * HD * LL / 256); // 65536

    for (int l = 0; l < LYR; l++) {
        const float* cur = (l == 0) ? x : out;
        ln512<<<MROWS, 256, 0, stream>>>(cur, ln1w + l * HD, ln1b + l * HD,
                                         P1, nullptr, P2);
        ssm_scan<<<scanBlocks, 256, 0, stream>>>(P2, dAr, dAi, cfr, cfi, l * HD * NHM);
        post_ssm<<<postBlocks, 256, 0, stream>>>(P2, P1, Dp, Q0, l);
        gemm512_wmma<<<gemmBlocks, 256, 0, stream>>>(Q0, WkT + l * HD * HD,
                                                     bk + l * HD, nullptr, nullptr, Q1, 0);
        gemm512_wmma<<<gemmBlocks, 256, 0, stream>>>(Q1, W1T + l * HD * HD,
                                                     b1 + l * HD, cur, P1, Q0, 0);
        ln512<<<MROWS, 256, 0, stream>>>(P1, ln2w + l * HD, ln2b + l * HD,
                                         nullptr, Q1, nullptr);
        gemm512_wmma<<<gemmBlocks, 256, 0, stream>>>(Q0, WpT + l * HD * HD,
                                                     bproj + l * HD, nullptr, P2, nullptr, 0);
        gemm512_wmma<<<gemmBlocks, 256, 0, stream>>>(Q1, W2T + l * HD * HD,
                                                     b2 + l * HD, nullptr, nullptr, Q0, 1);
        gemm512_wmma<<<gemmBlocks, 256, 0, stream>>>(Q0, WoT + l * HD * HD,
                                                     bout + l * HD, P2, out, nullptr, 0);
    }
    (void)in_sizes; (void)n_in; (void)out_size; (void)ws_size;
}
